// WaveNetBlock_73143293051424
// MI455X (gfx1250) — matrix operand order repaired and checked by reading, not compiled, not run on hardware
//
#include <hip/hip_runtime.h>
#include <hip/hip_bf16.h>
#include <math.h>

typedef __attribute__((ext_vector_type(8)))  float  v8f;
typedef __attribute__((ext_vector_type(8)))  __bf16 bf16x8;
typedef __attribute__((ext_vector_type(16))) __bf16 bf16x16;

#define H_    256
#define E_    128
#define B_    16
#define T_    16384
#define TOUT_ 16320
#define TILE_ 128          // time-tile per workgroup (last tile overlaps)
#define KD_   512          // fused conv K dim = H * 2 taps (memory order j = 2c+k)

#define XB_STR 520         // bf16 elements per XB row (16B-aligned stride)
#define F_STR  136         // bf16 elements per F row
#define ZT_STR 264         // bf16 elements per ZT row (16B-aligned stride)

#define XB_OFF 0
#define XB_BYTES (TILE_ * XB_STR * 2)        // 133120
#define F_OFF  XB_BYTES
#define F_BYTES (H_ * F_STR * 2)             // 69632
#define ZT_OFF (F_OFF + F_BYTES)             // 202752 (16B aligned)
#define ZT_BYTES (TILE_ * ZT_STR * 2)        // 67584
#define SMEM_BYTES (ZT_OFF + ZT_BYTES)       // 270336 B LDS (< 320KB WGP budget)

__device__ __forceinline__ unsigned short f2bf(float f) {
  unsigned int u = __float_as_uint(f);
  unsigned int r = u + 0x7FFFu + ((u >> 16) & 1u);   // round-to-nearest-even
  return (unsigned short)(r >> 16);
}

__device__ __forceinline__ float bf2f(unsigned short us) {
  return __uint_as_float((unsigned int)us << 16);
}

// Branch-free activations: v_exp_f32 + v_rcp_f32; saturate correctly via inf.
__device__ __forceinline__ float fast_tanh(float x) {
  float e = __expf(2.0f * x);                 // inf for large x, 0 for very neg
  return 1.0f - 2.0f * __builtin_amdgcn_rcpf(e + 1.0f);
}
__device__ __forceinline__ float fast_sig(float x) {
  return __builtin_amdgcn_rcpf(1.0f + __expf(-x));
}

__device__ __forceinline__ bf16x8 ld8(const unsigned short* p) {
  return *reinterpret_cast<const bf16x8*>(p);
}

__device__ __forceinline__ bf16x16 cat(bf16x8 a, bf16x8 b) {
  return __builtin_shufflevector(a, b, 0,1,2,3,4,5,6,7,8,9,10,11,12,13,14,15);
}

__device__ __forceinline__ v8f wmma_bf16(bf16x16 a, bf16x16 b, v8f c) {
  return __builtin_amdgcn_wmma_f32_16x16x32_bf16(false, a, false, b, (short)0, c, false, false);
}

// ---------------------------------------------------------------------------
// Prepack weights to bf16 in workspace. wf_dil/wg_dil [O,H,K] rows are already
// contiguous in the fused K index j=2c+k. Identity is folded into W_res.
// ---------------------------------------------------------------------------
__global__ void wn_prepack(const float* __restrict__ wf_dil,
                           const float* __restrict__ wg_dil,
                           const float* __restrict__ w_res,
                           const float* __restrict__ w_skip,
                           unsigned short* __restrict__ ws) {
  unsigned short* WfB = ws;                 // 256*512
  unsigned short* WgB = ws + 131072;        // 256*512
  unsigned short* WrB = ws + 262144;        // 256*256
  unsigned short* WsB = ws + 327680;        // 256*256
  for (int i = blockIdx.x * blockDim.x + threadIdx.x; i < 393216;
       i += gridDim.x * blockDim.x) {
    if (i < 131072) {
      WfB[i] = f2bf(wf_dil[i]);
    } else if (i < 262144) {
      int j = i - 131072; WgB[j] = f2bf(wg_dil[j]);
    } else if (i < 327680) {
      int j = i - 262144; int h = j >> 8, c = j & 255;
      WrB[j] = f2bf(w_res[j] + (h == c ? 1.0f : 0.0f));
    } else {
      int j = i - 327680; WsB[j] = f2bf(w_skip[j]);
    }
  }
}

// ---------------------------------------------------------------------------
// Conditioning projections (negligible FLOPs, plain VALU).
// ---------------------------------------------------------------------------
__global__ void wn_cond(const float* __restrict__ condition,
                        const float* __restrict__ wf_cond,
                        const float* __restrict__ bf_cond,
                        const float* __restrict__ wg_cond,
                        const float* __restrict__ bg_cond,
                        float* __restrict__ condf,
                        float* __restrict__ condg) {
  const int o = threadIdx.x;
  const float* W = (blockIdx.x == 0) ? wf_cond : wg_cond;
  const float* bias = (blockIdx.x == 0) ? bf_cond : bg_cond;
  float* outp = (blockIdx.x == 0) ? condf : condg;
  const float* wr = W + o * E_;
  for (int b = 0; b < B_; ++b) {
    float a = bias[o];
    const float* cb = condition + b * E_;
    for (int e = 0; e < E_; ++e) a = fmaf(wr[e], cb[e], a);
    outp[b * H_ + o] = a;
  }
}

// ---------------------------------------------------------------------------
// Fused main kernel: 512 threads (16 wave32) per (batch, 128-wide time tile).
// Waves 0-7: filt conv; waves 8-15: gate conv. Each wave owns a 64ch x 64t
// accumulator block (4x4 WMMA tiles). Gating goes through LDS (bf16 tanh),
// then all waves run the res/skip 256x256 GEMMs from bf16 z in LDS.
// ---------------------------------------------------------------------------
__global__ __launch_bounds__(512) void wn_main(
    const float* __restrict__ x,
    const unsigned short* __restrict__ WfB,
    const unsigned short* __restrict__ WgB,
    const unsigned short* __restrict__ WrB,
    const unsigned short* __restrict__ WsB,
    const float* __restrict__ condf,
    const float* __restrict__ condg,
    float* __restrict__ out) {
  extern __shared__ char smem[];
  unsigned short* XB = (unsigned short*)(smem + XB_OFF);  // [t][j=2c+k] bf16
  unsigned short* Fb = (unsigned short*)(smem + F_OFF);   // tanh(filt) [c][t] bf16
  unsigned short* ZT = (unsigned short*)(smem + ZT_OFF);  // z [t][c] bf16

  const int tid   = threadIdx.x;
  const int wave  = tid >> 5;    // 0..15
  const int lane  = tid & 31;
  const int lhalf = lane >> 4;   // 0|1 : half-wave (ISA fragment layout)
  const int l16   = lane & 15;

  const int blk = blockIdx.x;
  const int b   = blk >> 7;            // 16 batches
  const int tt  = blk & 127;           // 128 tiles per batch
  const int t0  = (tt == 127) ? (TOUT_ - TILE_) : tt * TILE_;  // overlap tail

  // ---- stage x[b, :, t0 .. t0+191] -> XB (bf16, transposed, tap-interleaved)
  {
    const float* xb = x + (size_t)b * H_ * T_ + t0;
    for (int c = wave; c < H_; c += 16) {
      const float* xc = xb + (size_t)c * T_;
      #pragma unroll
      for (int i = 0; i < 6; ++i) {
        const int t = lane + 32 * i;        // 0..191
        const unsigned short us = f2bf(xc[t]);
        if (t < TILE_) XB[t * XB_STR + 2 * c + 0] = us;           // tap k=0
        if (t >= 64)   XB[(t - 64) * XB_STR + 2 * c + 1] = us;    // tap k=1
      }
    }
  }
  __syncthreads();

  // ---- phase 1: dilated-conv GEMMs, acc initialized with conditioning bias
  const bool isF  = (wave < 8);
  const int  widx = isF ? wave : (wave - 8);
  const int  mo   = (widx & 3) * 64;        // channel block
  const int  nb   = (widx >> 2) * 64;       // time-column block (0 or 64)
  const unsigned short* WA   = isF ? WfB : WgB;
  const float*          cvec = isF ? condf : condg;

  v8f acc[4][4];
  #pragma unroll
  for (int mt = 0; mt < 4; ++mt) {
    v8f cv;
    const int rowb = mo + mt * 16 + lhalf * 8;
    #pragma unroll
    for (int r = 0; r < 8; ++r) cv[r] = cvec[b * H_ + rowb + r];
    #pragma unroll
    for (int nt = 0; nt < 4; ++nt) acc[mt][nt] = cv;
  }

  for (int j0 = 0; j0 < KD_; j0 += 32) {
    bf16x16 A[4];
    #pragma unroll
    for (int mt = 0; mt < 4; ++mt) {
      const unsigned short* ap =
          WA + (size_t)(mo + mt * 16 + l16) * KD_ + j0 + lhalf * 8;
      A[mt] = cat(ld8(ap), ld8(ap + 16));
    }
    #pragma unroll
    for (int nt = 0; nt < 4; ++nt) {
      const unsigned short* bp =
          XB + (nb + nt * 16 + l16) * XB_STR + j0 + lhalf * 8;
      bf16x16 Bf = cat(ld8(bp), ld8(bp + 16));
      #pragma unroll
      for (int mt = 0; mt < 4; ++mt)
        acc[mt][nt] = wmma_bf16(A[mt], Bf, acc[mt][nt]);
    }
  }

  // ---- gating: filt waves publish tanh (bf16); gate waves form z = sig*tanh
  if (isF) {
    #pragma unroll
    for (int mt = 0; mt < 4; ++mt)
      #pragma unroll
      for (int nt = 0; nt < 4; ++nt)
        #pragma unroll
        for (int r = 0; r < 8; ++r) {
          const int c = mo + mt * 16 + lhalf * 8 + r;
          const int t = nb + nt * 16 + l16;
          Fb[c * F_STR + t] = f2bf(fast_tanh(acc[mt][nt][r]));
        }
  }
  __syncthreads();
  if (!isF) {
    #pragma unroll
    for (int mt = 0; mt < 4; ++mt)
      #pragma unroll
      for (int nt = 0; nt < 4; ++nt)
        #pragma unroll
        for (int r = 0; r < 8; ++r) {
          const int c = mo + mt * 16 + lhalf * 8 + r;
          const int t = nb + nt * 16 + l16;
          const float z = fast_sig(acc[mt][nt][r]) * bf2f(Fb[c * F_STR + t]);
          ZT[t * ZT_STR + c] = f2bf(z);
        }
  }
  __syncthreads();

  // ---- phase 3: residual (waves 0-7, identity folded) / skip (waves 8-15)
  const unsigned short* WP = isF ? WrB : WsB;
  #pragma unroll
  for (int mt = 0; mt < 4; ++mt)
    #pragma unroll
    for (int nt = 0; nt < 4; ++nt) {
      v8f z = {};
      acc[mt][nt] = z;
    }

  for (int k0 = 0; k0 < H_; k0 += 32) {
    bf16x16 A[4];
    #pragma unroll
    for (int mt = 0; mt < 4; ++mt) {
      const unsigned short* ap =
          WP + (size_t)(mo + mt * 16 + l16) * H_ + k0 + lhalf * 8;
      A[mt] = cat(ld8(ap), ld8(ap + 16));
    }
    #pragma unroll
    for (int nt = 0; nt < 4; ++nt) {
      const unsigned short* bp =
          ZT + (nb + nt * 16 + l16) * ZT_STR + k0 + lhalf * 8;
      bf16x16 Bf = cat(ld8(bp), ld8(bp + 16));
      #pragma unroll
      for (int mt = 0; mt < 4; ++mt)
        acc[mt][nt] = wmma_bf16(A[mt], Bf, acc[mt][nt]);
    }
  }

  float* outp = out + (isF ? (size_t)0 : (size_t)B_ * H_ * TOUT_);
  #pragma unroll
  for (int mt = 0; mt < 4; ++mt)
    #pragma unroll
    for (int nt = 0; nt < 4; ++nt)
      #pragma unroll
      for (int r = 0; r < 8; ++r) {
        const int hh = mo + mt * 16 + lhalf * 8 + r;
        const int tc = t0 + nb + nt * 16 + l16;
        outp[((size_t)b * H_ + hh) * TOUT_ + tc] = acc[mt][nt][r];
      }
}

extern "C" void kernel_launch(void* const* d_in, const int* in_sizes, int n_in,
                              void* d_out, int out_size, void* d_ws, size_t ws_size,
                              hipStream_t stream) {
  const float* x         = (const float*)d_in[0];
  const float* condition = (const float*)d_in[1];
  const float* wf_dil    = (const float*)d_in[2];
  const float* wf_cond   = (const float*)d_in[3];
  const float* bf_cond   = (const float*)d_in[4];
  const float* wg_dil    = (const float*)d_in[5];
  const float* wg_cond   = (const float*)d_in[6];
  const float* bg_cond   = (const float*)d_in[7];
  const float* w_res     = (const float*)d_in[8];
  const float* w_skip    = (const float*)d_in[9];

  unsigned short* wsu = (unsigned short*)d_ws;
  unsigned short* WfB = wsu;                       // 256*512 bf16
  unsigned short* WgB = wsu + 131072;              // 256*512 bf16
  unsigned short* WrB = wsu + 262144;              // 256*256 bf16 (+I folded)
  unsigned short* WsB = wsu + 327680;              // 256*256 bf16
  float* condf = (float*)(wsu + 393216);           // 16*256 f32
  float* condg = condf + B_ * H_;                  // 16*256 f32

  wn_prepack<<<512, 256, 0, stream>>>(wf_dil, wg_dil, w_res, w_skip, wsu);
  wn_cond<<<2, 256, 0, stream>>>(condition, wf_cond, bf_cond, wg_cond, bg_cond,
                                 condf, condg);

  (void)hipFuncSetAttribute((const void*)wn_main,
                            hipFuncAttributeMaxDynamicSharedMemorySize,
                            SMEM_BYTES);
  wn_main<<<B_ * 128, 512, SMEM_BYTES, stream>>>(
      x, WfB, WgB, WrB, WsB, condf, condg, (float*)d_out);
}
